// GPT_30459908063443
// MI455X (gfx1250) — matrix operand order repaired and checked by reading, not compiled, not run on hardware
//
#include <hip/hip_runtime.h>
#include <hip/hip_bf16.h>
#include <math.h>

typedef __attribute__((ext_vector_type(16))) _Float16 v16h;
typedef __attribute__((ext_vector_type(4)))  _Float16 h4;
typedef __attribute__((ext_vector_type(8)))  float    v8f;

constexpr int Cc = 768, Tt = 1024, Hh = 12, Nh = 64, FF = 3072, Vv = 50257;
constexpr int TCn = Tt * Cc;   // 786432

// ======================= WMMA GEMM =======================
// O[M,N] = epi( A[M,K] @ B )  where B = W^T (WT=1, W is [N,K] row-major)
//                               or  B = W   (WT=0, W is [K,N] row-major)
// Block: 256 threads = 8 waves; block tile 128(M) x 64(N); K step 32.
// Waves arranged 4(M) x 2(N); each wave owns a 32x32 output quadrant:
// 2 A frags x 2 B frags -> 4x v_wmma_f32_16x16x32_f16 per wave per K step.
// GUARD=0 requires M%128==0, N%64==0, K%32==0, 16B-aligned rows.
template<int EPI, int WT, int GUARD>
__global__ __launch_bounds__(256) void k_gemm(
    const float* __restrict__ A, int lda,
    const float* __restrict__ W, int ldw,
    float* __restrict__ O, int ldo, int M, int N, int K)
{
  __shared__ _Float16 As[128][40];  // [m][k], +8 half pad
  __shared__ _Float16 Bs[64][40];   // [n][k] (so K-pairs are contiguous)
  const int tid  = threadIdx.x;
  const int lane = tid & 31;
  const int wave = tid >> 5;
  const int wm = wave & 3;                       // M group (32 rows)
  const int wn = wave >> 2;                      // N group (32 cols)
  const int m0 = blockIdx.y * 128;
  const int n0 = blockIdx.x * 64;
  const int mrow = lane & 15;
  const int hw   = lane >> 4;                    // lane half (wave32)
  v8f acc00 = {}, acc01 = {}, acc10 = {}, acc11 = {};
  for (int k0 = 0; k0 < K; k0 += 32) {
    if (GUARD) {
      // scalar staging with bounds checks (only small/tail GEMMs)
#pragma unroll
      for (int i = 0; i < 16; ++i) {             // A: 128x32
        int id = tid + i * 256;
        int r = id >> 5, kk = id & 31;
        int kg = k0 + kk;
        float va = (kg < K) ? A[(size_t)(m0 + r) * lda + kg] : 0.f;
        As[r][kk] = (_Float16)va;
      }
#pragma unroll
      for (int i = 0; i < 8; ++i) {              // B: 64x32
        int id = tid + i * 256;
        int r = id >> 5, kk = id & 31;
        int kg = k0 + kk, ng = n0 + r;
        float vb = 0.f;
        if (ng < N && kg < K)
          vb = WT ? W[(size_t)ng * ldw + kg] : W[(size_t)kg * ldw + ng];
        Bs[r][kk] = (_Float16)vb;
      }
    } else {
      // vectorized staging: float4 global loads -> packed half4 LDS stores
#pragma unroll
      for (int i = 0; i < 4; ++i) {              // A: 128 rows x 8 chunks
        int id = tid + i * 256;                  // 0..1023
        int r = id >> 3;                         // 0..127
        int kc = (id & 7) * 4;                   // 0,4,...,28
        const float4 f = *reinterpret_cast<const float4*>(
            &A[(size_t)(m0 + r) * lda + k0 + kc]);
        *reinterpret_cast<h4*>(&As[r][kc]) =
            (h4){(_Float16)f.x, (_Float16)f.y, (_Float16)f.z, (_Float16)f.w};
      }
#pragma unroll
      for (int i = 0; i < 2; ++i) {              // B: 64 rows x 8 chunks
        int id = tid + i * 256;                  // 0..511
        int r = id >> 3;                         // 0..63
        int kc = (id & 7) * 4;
        if (WT) {
          const float4 g = *reinterpret_cast<const float4*>(
              &W[(size_t)(n0 + r) * ldw + k0 + kc]);
          *reinterpret_cast<h4*>(&Bs[r][kc]) =
              (h4){(_Float16)g.x, (_Float16)g.y, (_Float16)g.z, (_Float16)g.w};
        } else {
          const float* wp = &W[(size_t)(k0 + kc) * ldw + n0 + r];
          *reinterpret_cast<h4*>(&Bs[r][kc]) =
              (h4){(_Float16)wp[0], (_Float16)wp[(size_t)ldw],
                   (_Float16)wp[2 * (size_t)ldw], (_Float16)wp[3 * (size_t)ldw]};
        }
      }
    }
    if (k0 + 32 < K)                             // CDNA5 global_prefetch_b8
      __builtin_prefetch(&A[(size_t)(m0 + wave * 16) * lda + k0 + 32], 0, 1);
    __syncthreads();
    // B fragments: 32x16 f16. VGPR j<4: K=2j+hw*8 ; j>=4: K=16+2(j-4)+hw*8
    v16h b0, b1;
#pragma unroll
    for (int j = 0; j < 8; ++j) {
      int kb = (j < 4 ? 2 * j : 16 + 2 * (j - 4)) + hw * 8;
      b0[2 * j]     = Bs[wn * 32 + mrow][kb];
      b0[2 * j + 1] = Bs[wn * 32 + mrow][kb + 1];
      b1[2 * j]     = Bs[wn * 32 + 16 + mrow][kb];
      b1[2 * j + 1] = Bs[wn * 32 + 16 + mrow][kb + 1];
    }
    // A fragments: 16x32 f16. VGPR j<4: K=2j+hw*8 ; j>=4: K=16+2(j-4)+hw*8
    v16h a0, a1;
#pragma unroll
    for (int j = 0; j < 8; ++j) {
      int kb = (j < 4 ? 2 * j : 16 + 2 * (j - 4)) + hw * 8;
      a0[2 * j]     = As[wm * 32 + mrow][kb];
      a0[2 * j + 1] = As[wm * 32 + mrow][kb + 1];
      a1[2 * j]     = As[wm * 32 + 16 + mrow][kb];
      a1[2 * j + 1] = As[wm * 32 + 16 + mrow][kb + 1];
    }
    acc00 = __builtin_amdgcn_wmma_f32_16x16x32_f16(
        false, a0, false, b0, (short)0, acc00, false, false);
    acc01 = __builtin_amdgcn_wmma_f32_16x16x32_f16(
        false, a0, false, b1, (short)0, acc01, false, false);
    acc10 = __builtin_amdgcn_wmma_f32_16x16x32_f16(
        false, a1, false, b0, (short)0, acc10, false, false);
    acc11 = __builtin_amdgcn_wmma_f32_16x16x32_f16(
        false, a1, false, b1, (short)0, acc11, false, false);
    __syncthreads();
  }
  // C/D layout: VGPR v: lanes0-15 -> M=v, lanes16-31 -> M=v+8; N = lane&15
#pragma unroll
  for (int ia = 0; ia < 2; ++ia) {
#pragma unroll
    for (int ib = 0; ib < 2; ++ib) {
      v8f acc = ia ? (ib ? acc11 : acc10) : (ib ? acc01 : acc00);
#pragma unroll
      for (int vv = 0; vv < 8; ++vv) {
        int row = m0 + wm * 32 + ia * 16 + vv + hw * 8;
        int col = n0 + wn * 32 + ib * 16 + mrow;
        float x = acc[vv];
        if (EPI == 1)      x = tanhf(x);
        else if (EPI == 2) x = 1.f / (1.f + expf(-x));
        else if (EPI == 3) { x = fmaxf(x, 0.f); x = x * x; }
        if (GUARD) {
          if (row < M && col < N) O[(size_t)row * ldo + col] = x;
        } else {
          O[(size_t)row * ldo + col] = x;
        }
      }
    }
  }
}

// ======================= LayerNorm (one block / token) =======================
__global__ __launch_bounds__(256) void k_ln(
    const float* __restrict__ X, const float* __restrict__ g,
    const float* __restrict__ b, float* __restrict__ O)
{
  const int t = blockIdx.x, tid = threadIdx.x;
  __shared__ float red[8], red2[8];
  float v[3]; float s = 0.f;
#pragma unroll
  for (int i = 0; i < 3; ++i) { v[i] = X[(size_t)t * Cc + tid + i * 256]; s += v[i]; }
  for (int m = 16; m >= 1; m >>= 1) s += __shfl_xor(s, m);
  if ((tid & 31) == 0) red[tid >> 5] = s;
  __syncthreads();
  float mu = 0.f;
#pragma unroll
  for (int i = 0; i < 8; ++i) mu += red[i];
  mu *= (1.f / Cc);
  float vs = 0.f;
#pragma unroll
  for (int i = 0; i < 3; ++i) { float d = v[i] - mu; vs += d * d; }
  for (int m = 16; m >= 1; m >>= 1) vs += __shfl_xor(vs, m);
  if ((tid & 31) == 0) red2[tid >> 5] = vs;
  __syncthreads();
  float var = 0.f;
#pragma unroll
  for (int i = 0; i < 8; ++i) var += red2[i];
  float rs = rsqrtf(var * (1.f / Cc) + 1e-5f);
#pragma unroll
  for (int i = 0; i < 3; ++i) {
    int c = tid + i * 256;
    O[(size_t)t * Cc + c] = (v[i] - mu) * rs * g[c] + b[c];
  }
}

// Embedding gather + ln0 fused
__global__ __launch_bounds__(256) void k_embed_ln(
    const float* __restrict__ emb, const int* __restrict__ idx,
    const float* __restrict__ g, const float* __restrict__ b,
    float* __restrict__ O)
{
  const int t = blockIdx.x, tid = threadIdx.x;
  const size_t rb = (size_t)idx[t] * Cc;
  __shared__ float red[8], red2[8];
  float v[3]; float s = 0.f;
#pragma unroll
  for (int i = 0; i < 3; ++i) { v[i] = emb[rb + tid + i * 256]; s += v[i]; }
  for (int m = 16; m >= 1; m >>= 1) s += __shfl_xor(s, m);
  if ((tid & 31) == 0) red[tid >> 5] = s;
  __syncthreads();
  float mu = 0.f;
#pragma unroll
  for (int i = 0; i < 8; ++i) mu += red[i];
  mu *= (1.f / Cc);
  float vs = 0.f;
#pragma unroll
  for (int i = 0; i < 3; ++i) { float d = v[i] - mu; vs += d * d; }
  for (int m = 16; m >= 1; m >>= 1) vs += __shfl_xor(vs, m);
  if ((tid & 31) == 0) red2[tid >> 5] = vs;
  __syncthreads();
  float var = 0.f;
#pragma unroll
  for (int i = 0; i < 8; ++i) var += red2[i];
  float rs = rsqrtf(var * (1.f / Cc) + 1e-5f);
#pragma unroll
  for (int i = 0; i < 3; ++i) {
    int c = tid + i * 256;
    O[(size_t)t * Cc + c] = (v[i] - mu) * rs * g[c] + b[c];
  }
}

// ======================= elementwise =======================
#define GS_LOOP for (int i = blockIdx.x * blockDim.x + threadIdx.x; i < TCn; i += gridDim.x * blockDim.x)

__global__ void k_tokshift_mix1(const float* __restrict__ xn,
                                const float* __restrict__ coef,
                                float* __restrict__ out)
{
  GS_LOOP {
    int t = i / Cc, c = i - t * Cc;
    float x = xn[i];
    float xx = (t ? xn[i - Cc] : 0.f) - x;
    out[i] = x + xx * coef[c];
  }
}

__global__ void k_six_mix(const float* __restrict__ xn, float* __restrict__ m6,
                          const float* __restrict__ vr, const float* __restrict__ vw,
                          const float* __restrict__ vk, const float* __restrict__ vv,
                          const float* __restrict__ va, const float* __restrict__ vg)
{
  GS_LOOP {
    int t = i / Cc, c = i - t * Cc;
    float x = xn[i];
    float xx = (t ? xn[i - Cc] : 0.f) - x;
    m6[i]            = x + xx * (vr[c] + m6[i]);
    m6[i + 1 * TCn]  = x + xx * (vw[c] + m6[i + 1 * TCn]);
    m6[i + 2 * TCn]  = x + xx * (vk[c] + m6[i + 2 * TCn]);
    m6[i + 3 * TCn]  = x + xx * (vv[c] + m6[i + 3 * TCn]);
    m6[i + 4 * TCn]  = x + xx * (va[c] + m6[i + 4 * TCn]);
    m6[i + 5 * TCn]  = x + xx * (vg[c] + m6[i + 5 * TCn]);
  }
}

__global__ void k_wfinal(const float* __restrict__ dec0, const float* __restrict__ lin,
                         float* __restrict__ Wout, float* __restrict__ EWout)
{
  GS_LOOP {
    int c = i % Cc;
    float u = dec0[c] + lin[i];
    float w = fminf(u, 0.f) - log1pf(expf(-fabsf(u))) - 0.5f;
    Wout[i] = w;
    EWout[i] = expf(w);
  }
}

__global__ void k_bias_sig(const float* __restrict__ bias, const float* __restrict__ lin,
                           float* __restrict__ out)
{
  GS_LOOP { int c = i % Cc; out[i] = 1.f / (1.f + expf(-(bias[c] + lin[i]))); }
}

__global__ void k_kmod(float* __restrict__ K_, const float* __restrict__ A_,
                       const float* __restrict__ MA, const float* __restrict__ MK,
                       const float* __restrict__ W_)
{
  GS_LOOP {
    float kv = K_[i], m = MA[i];
    float kn = kv * m + kv * A_[i] * (1.f - m);
    K_[i] = kn * expf(fminf(W_[i] * MK[i], 0.f));
  }
}

__global__ void k_ab(const float* __restrict__ KK, const float* __restrict__ A_,
                     float* __restrict__ AA, float* __restrict__ BB)
{
  GS_LOOP { float kv = KK[i]; AA[i] = -kv; BB[i] = kv * A_[i]; }
}

__global__ void k_mul(const float* __restrict__ a, const float* __restrict__ b,
                      float* __restrict__ o)
{ GS_LOOP o[i] = a[i] * b[i]; }

__global__ void k_add(float* __restrict__ x, const float* __restrict__ y)
{ GS_LOOP x[i] += y[i]; }

__global__ void k_muladd(float* __restrict__ x, const float* __restrict__ a,
                         const float* __restrict__ b)
{ GS_LOOP x[i] += a[i] * b[i]; }

// ======================= per-(t,h) wave32 reductions =======================
__global__ void k_kk_norm(const float* __restrict__ K_, const float* __restrict__ KKD,
                          float* __restrict__ KK)
{
  int th = blockIdx.x;
  int t = th / Hh, h = th - t * Hh;
  int lane = threadIdx.x;
  size_t off = (size_t)t * Cc + h * Nh;
  float v0 = K_[off + lane]      + KKD[off + lane];
  float v1 = K_[off + 32 + lane] + KKD[off + 32 + lane];
  float ss = v0 * v0 + v1 * v1;
  for (int m = 16; m >= 1; m >>= 1) ss += __shfl_xor(ss, m);
  float nrm = fmaxf(sqrtf(ss), 1e-12f);
  KK[off + lane]      = v0 / nrm;
  KK[off + 32 + lane] = v1 / nrm;
}

__global__ void k_gn_bonus(float* __restrict__ Y, const float* __restrict__ R_,
                           const float* __restrict__ K_, const float* __restrict__ V_,
                           const float* __restrict__ faaaa,
                           const float* __restrict__ gg, const float* __restrict__ gb)
{
  int th = blockIdx.x;
  int t = th / Hh, h = th - t * Hh;
  int lane = threadIdx.x;
  size_t off = (size_t)t * Cc + h * Nh;
  float y0 = Y[off + lane], y1 = Y[off + 32 + lane];
  float s = y0 + y1;
  for (int m = 16; m >= 1; m >>= 1) s += __shfl_xor(s, m);
  float mu = s * (1.f / Nh);
  float d0 = y0 - mu, d1 = y1 - mu;
  float vs = d0 * d0 + d1 * d1;
  for (int m = 16; m >= 1; m >>= 1) vs += __shfl_xor(vs, m);
  float rstd = rsqrtf(vs * (1.f / Nh) + 6.4e-4f);
  int c0 = h * Nh + lane, c1 = c0 + 32;
  y0 = d0 * rstd * gg[c0] + gb[c0];
  y1 = d1 * rstd * gg[c1] + gb[c1];
  float dot = R_[off + lane] * K_[off + lane] * faaaa[c0]
            + R_[off + 32 + lane] * K_[off + 32 + lane] * faaaa[c1];
  for (int m = 16; m >= 1; m >>= 1) dot += __shfl_xor(dot, m);
  Y[off + lane]      = y0 + dot * V_[off + lane];
  Y[off + 32 + lane] = y1 + dot * V_[off + 32 + lane];
}

// ======================= WKV7 recurrence =======================
__global__ __launch_bounds__(256) void k_wkv(
    const float* __restrict__ R_, const float* __restrict__ EW,
    const float* __restrict__ K_, const float* __restrict__ V_,
    const float* __restrict__ AA, const float* __restrict__ BB,
    float* __restrict__ Y)
{
  const int h = blockIdx.x;
  const int tid = threadIdx.x;
  const int row = tid >> 2;
  const int cq = tid & 3;
  const int j0 = cq * 16;
  float S[16];
#pragma unroll
  for (int i = 0; i < 16; ++i) S[i] = 0.f;
  __shared__ float sh[6 * Nh];
  float* shr = sh;            float* shw = sh + Nh;     float* shk = sh + 2 * Nh;
  float* shv = sh + 3 * Nh;   float* sha = sh + 4 * Nh; float* shb = sh + 5 * Nh;
  const size_t cbase = (size_t)h * Nh;
  for (int t = 0; t < Tt; ++t) {
    size_t off = (size_t)t * Cc + cbase;
    if (tid < 64) {
      shr[tid] = R_[off + tid];
      shw[tid] = EW[off + tid];
      shk[tid] = K_[off + tid];
    } else if (tid < 128) {
      int u = tid - 64;
      shv[u] = V_[off + u];
      sha[u] = AA[off + u];
      shb[u] = BB[off + u];
    }
    __syncthreads();
    float sa = 0.f;
#pragma unroll
    for (int i = 0; i < 16; ++i) sa += S[i] * sha[j0 + i];
    sa += __shfl_xor(sa, 1);
    sa += __shfl_xor(sa, 2);
    float vr = shv[row];
    float y = 0.f;
#pragma unroll
    for (int i = 0; i < 16; ++i) {
      int j = j0 + i;
      S[i] = S[i] * shw[j] + sa * shb[j] + vr * shk[j];
      y += S[i] * shr[j];
    }
    y += __shfl_xor(y, 1);
    y += __shfl_xor(y, 2);
    if (cq == 0) Y[off + row] = y;
    __syncthreads();
  }
}

// ======================= host orchestration =======================
enum {
  P_MAA_X = 0, P_MAA_R, P_MAA_W, P_MAA_K, P_MAA_V, P_MAA_A, P_MAA_G,
  P_AAAAA, P_MISC_K, P_MISC_A, P_C_MAA_K, P_C_MAA_R,
  P_DECAY0, P_FAAAA, P_MAA_W1, P_MAA_W2,
  P_DEC_W1, P_DEC_W2, P_AAA_W1, P_AAA_W2, P_KKK_W1, P_KKK_W2,
  P_GATE_W1, P_GATE_W2, P_MK_W1, P_MK_W2, P_MA_W1, P_MA_W2,
  P_WR, P_WK, P_WV, P_WO,
  P_GN_G, P_GN_B, P_LN1_G, P_LN1_B, P_LN2_G, P_LN2_B,
  P_CW_K, P_CW_R, P_CW_V, P_NPER
};

template<int EPI>
static void gemm_e(hipStream_t s, const float* A, int lda,
                   const float* W, int ldw, int wT,
                   float* O, int ldo, int M, int N, int K)
{
  dim3 g((N + 63) / 64, (M + 127) / 128);
  bool guard = (N & 63) || (K & 31);
  if (wT) {
    if (guard) k_gemm<EPI,1,1><<<g,256,0,s>>>(A,lda,W,ldw,O,ldo,M,N,K);
    else       k_gemm<EPI,1,0><<<g,256,0,s>>>(A,lda,W,ldw,O,ldo,M,N,K);
  } else {
    if (guard) k_gemm<EPI,0,1><<<g,256,0,s>>>(A,lda,W,ldw,O,ldo,M,N,K);
    else       k_gemm<EPI,0,0><<<g,256,0,s>>>(A,lda,W,ldw,O,ldo,M,N,K);
  }
}

static void gemm(hipStream_t s, int epi,
                 const float* A, int lda, const float* W, int ldw, int wT,
                 float* O, int ldo, int M, int N, int K)
{
  switch (epi) {
    case 0: gemm_e<0>(s,A,lda,W,ldw,wT,O,ldo,M,N,K); break;
    case 1: gemm_e<1>(s,A,lda,W,ldw,wT,O,ldo,M,N,K); break;
    case 2: gemm_e<2>(s,A,lda,W,ldw,wT,O,ldo,M,N,K); break;
    default: gemm_e<3>(s,A,lda,W,ldw,wT,O,ldo,M,N,K); break;
  }
}

extern "C" void kernel_launch(void* const* d_in, const int* in_sizes, int n_in,
                              void* d_out, int out_size, void* d_ws, size_t ws_size,
                              hipStream_t stream)
{
  (void)in_sizes; (void)n_in; (void)out_size; (void)ws_size;
  const int NL = 6;
  const int G0 = NL * P_NPER; // 246
  auto LP = [&](int l, int j) { return (const float*)d_in[l * P_NPER + j]; };
  const float* emb    = (const float*)d_in[G0 + 0];
  const float* head   = (const float*)d_in[G0 + 1];
  const float* ln0g   = (const float*)d_in[G0 + 2];
  const float* ln0b   = (const float*)d_in[G0 + 3];
  const float* lnog   = (const float*)d_in[G0 + 4];
  const float* lnob   = (const float*)d_in[G0 + 5];
  const int*   idx    = (const int*)  d_in[G0 + 6];

  float* ws = (float*)d_ws;
  const size_t S = (size_t)TCn;
  float* X    = ws;               // residual
  float* XN   = ws + 1 * S;       // layernorm output
  float* SCR  = ws + 2 * S;       // xxx / xk2
  float* M6   = ws + 3 * S;       // 6 slots: m then xr..xg
  float* R_   = ws + 9 * S;
  float* W_   = ws + 10 * S;
  float* K_   = ws + 11 * S;
  float* V_   = ws + 12 * S;
  float* G_   = ws + 13 * S;
  float* KK_  = ws + 14 * S;
  float* A_   = ws + 15 * S;      // sigmoid a
  float* EW_  = ws + 16 * S;      // exp(w)
  float* AA_  = ws + 17 * S;      // -kk
  float* BB_  = ws + 18 * S;      // kk*a
  float* Y_   = ws + 19 * S;
  float* TMP  = ws + 20 * S;      // generic [T,C]
  float* MA_  = ws + 21 * S;
  float* SIG_ = ws + 22 * S;
  float* H192 = ws + 23 * S;      // [T,192]
  float* HS   = ws + 24 * S;      // [T,<=128] hidden
  float* KF   = ws + 25 * S;      // [T,FFN] (4*S)
  float* MK_  = ws + 29 * S;

  const dim3 EWG(768), EWB(256);
  const int THB = Tt * Hh;

  k_embed_ln<<<Tt, 256, 0, stream>>>(emb, idx, ln0g, ln0b, X);

  for (int l = 0; l < NL; ++l) {
    // ---------------- tmix ----------------
    k_ln<<<Tt, 256, 0, stream>>>(X, LP(l, P_LN1_G), LP(l, P_LN1_B), XN);
    k_tokshift_mix1<<<EWG, EWB, 0, stream>>>(XN, LP(l, P_MAA_X), SCR);
    gemm(stream, 1, SCR, Cc, LP(l, P_MAA_W1), 192, 0, H192, 192, Tt, 192, Cc); // tanh
    for (int j = 0; j < 6; ++j)
      gemm(stream, 0, H192 + j * 32, 192, LP(l, P_MAA_W2) + (size_t)j * 32 * Cc, Cc, 0,
           M6 + (size_t)j * S, Cc, Tt, Cc, 32);
    k_six_mix<<<EWG, EWB, 0, stream>>>(XN, M6, LP(l, P_MAA_R), LP(l, P_MAA_W),
                                       LP(l, P_MAA_K), LP(l, P_MAA_V),
                                       LP(l, P_MAA_A), LP(l, P_MAA_G));
    float* xr = M6;               float* xw = M6 + 1 * S;
    float* xk = M6 + 2 * S;       float* xv = M6 + 3 * S;
    float* xa = M6 + 4 * S;       float* xg = M6 + 5 * S;

    gemm(stream, 0, xr, Cc, LP(l, P_WR), Cc, 1, R_, Cc, Tt, Cc, Cc);
    // decay
    gemm(stream, 1, xw, Cc, LP(l, P_DEC_W1), 64, 0, HS, 64, Tt, 64, Cc);
    gemm(stream, 0, HS, 64, LP(l, P_DEC_W2), Cc, 0, TMP, Cc, Tt, Cc, 64);
    k_wfinal<<<EWG, EWB, 0, stream>>>(LP(l, P_DECAY0), TMP, W_, EW_);
    // k, v
    gemm(stream, 0, xk, Cc, LP(l, P_WK), Cc, 1, K_, Cc, Tt, Cc, Cc);
    gemm(stream, 0, xv, Cc, LP(l, P_WV), Cc, 1, V_, Cc, Tt, Cc, Cc);
    // gate
    gemm(stream, 1, xg, Cc, LP(l, P_GATE_W1), 128, 0, HS, 128, Tt, 128, Cc);
    gemm(stream, 0, HS, 128, LP(l, P_GATE_W2), Cc, 0, G_, Cc, Tt, Cc, 128);
    // kk
    gemm(stream, 1, xk, Cc, LP(l, P_KKK_W1), 64, 0, HS, 64, Tt, 64, Cc);
    gemm(stream, 0, HS, 64, LP(l, P_KKK_W2), Cc, 0, TMP, Cc, Tt, Cc, 64);
    k_kk_norm<<<THB, 32, 0, stream>>>(K_, TMP, KK_);
    // a
    gemm(stream, 0, xa, Cc, LP(l, P_AAA_W1), 64, 0, HS, 64, Tt, 64, Cc);
    gemm(stream, 0, HS, 64, LP(l, P_AAA_W2), Cc, 0, TMP, Cc, Tt, Cc, 64);
    k_bias_sig<<<EWG, EWB, 0, stream>>>(LP(l, P_AAAAA), TMP, A_);
    // ma
    gemm(stream, 0, xa, Cc, LP(l, P_MA_W1), 16, 0, HS, 16, Tt, 16, Cc);
    gemm(stream, 0, HS, 16, LP(l, P_MA_W2), Cc, 0, TMP, Cc, Tt, Cc, 16);
    k_bias_sig<<<EWG, EWB, 0, stream>>>(LP(l, P_MISC_A), TMP, MA_);
    // mk
    gemm(stream, 0, xk, Cc, LP(l, P_MK_W1), 16, 0, HS, 16, Tt, 16, Cc);
    gemm(stream, 0, HS, 16, LP(l, P_MK_W2), Cc, 0, TMP, Cc, Tt, Cc, 16);
    k_bias_sig<<<EWG, EWB, 0, stream>>>(LP(l, P_MISC_K), TMP, MK_);
    // modify k, build a/b, run WKV
    k_kmod<<<EWG, EWB, 0, stream>>>(K_, A_, MA_, MK_, W_);
    k_ab<<<EWG, EWB, 0, stream>>>(KK_, A_, AA_, BB_);
    k_wkv<<<Hh, 256, 0, stream>>>(R_, EW_, K_, V_, AA_, BB_, Y_);
    k_gn_bonus<<<THB, 32, 0, stream>>>(Y_, R_, K_, V_, LP(l, P_FAAAA),
                                       LP(l, P_GN_G), LP(l, P_GN_B));
    // out proj + residual
    k_mul<<<EWG, EWB, 0, stream>>>(Y_, G_, TMP);
    gemm(stream, 0, TMP, Cc, LP(l, P_WO), Cc, 1, Y_, Cc, Tt, Cc, Cc);
    k_add<<<EWG, EWB, 0, stream>>>(X, Y_);

    // ---------------- cmix ----------------
    k_ln<<<Tt, 256, 0, stream>>>(X, LP(l, P_LN2_G), LP(l, P_LN2_B), XN);
    k_tokshift_mix1<<<EWG, EWB, 0, stream>>>(XN, LP(l, P_C_MAA_K), SCR);
    k_tokshift_mix1<<<EWG, EWB, 0, stream>>>(XN, LP(l, P_C_MAA_R), TMP);
    gemm(stream, 3, SCR, Cc, LP(l, P_CW_K), Cc, 1, KF, FF, Tt, FF, Cc);   // relu^2
    gemm(stream, 2, TMP, Cc, LP(l, P_CW_R), Cc, 1, SIG_, Cc, Tt, Cc, Cc); // sigmoid
    gemm(stream, 0, KF, FF, LP(l, P_CW_V), FF, 1, Y_, Cc, Tt, Cc, FF);
    k_muladd<<<EWG, EWB, 0, stream>>>(X, SIG_, Y_);
  }

  // final LN + head
  k_ln<<<Tt, 256, 0, stream>>>(X, lnog, lnob, XN);
  gemm(stream, 0, XN, Cc, head, Cc, 1, (float*)d_out, Vv, Tt, Vv, Cc);
}
